// TemporalGraphAttention_61151744360724
// MI455X (gfx1250) — compile-verified, matrix operand order checked by reading, not run on hardware
//
#include <hip/hip_runtime.h>

// MI455X / gfx1250 fused multi-head attention, f16 WMMA path with f32 accumulate.
// B=4, S=2048, D=512, H=8, HD=64.
// Round 3: identical device code to round 2 (wmma + async-to-LDS pipeline);
// removes a duplicated Wv convert launch from the host side.

typedef __attribute__((ext_vector_type(16))) _Float16 v16h;
typedef __attribute__((ext_vector_type(8)))  _Float16 v8h;
typedef __attribute__((ext_vector_type(8)))  float    v8f;

union FragU { v16h v; v8h h[2]; };

// Load one 16x32 f16 WMMA operand fragment (A layout; B uses the same packing
// with "row" = B-column) from a row-major matrix with pitch `pitch` (halves).
// Lane l (l&15 = row, l>>4 = g) holds K = {8g..8g+7, 16+8g..23+8g} per ISA
// 16-bit A-matrix table -> two contiguous 16-byte loads.
__device__ __forceinline__ v16h load_frag(const _Float16* base, int pitch,
                                          int row0, int k0, int lane) {
  const int g = lane >> 4;
  const int r = lane & 15;
  const _Float16* p = base + (size_t)(row0 + r) * (size_t)pitch + (k0 + 8 * g);
  FragU f;
  f.h[0] = *(const v8h*)(p);        // K = k0 + 8g .. +7
  f.h[1] = *(const v8h*)(p + 16);   // K = k0 + 16 + 8g .. +7
  return f.v;
}

__device__ __forceinline__ v8f wmma16(v16h a, v16h b, v8f c) {
  return __builtin_amdgcn_wmma_f32_16x16x32_f16(false, a, false, b,
                                                (short)0, c, false, false);
}

__device__ __forceinline__ float rmax16(float v) {
  v = fmaxf(v, __shfl_xor(v, 1, 32));
  v = fmaxf(v, __shfl_xor(v, 2, 32));
  v = fmaxf(v, __shfl_xor(v, 4, 32));
  v = fmaxf(v, __shfl_xor(v, 8, 32));
  return v;
}
__device__ __forceinline__ float rsum16(float v) {
  v += __shfl_xor(v, 1, 32);
  v += __shfl_xor(v, 2, 32);
  v += __shfl_xor(v, 4, 32);
  v += __shfl_xor(v, 8, 32);
  return v;
}

// One 16-byte async global->LDS copy (per-lane). ASYNCcnt-tracked.
__device__ __forceinline__ void async_b128(const _Float16* g,
                                           const _Float16* l) {
  const unsigned loff = (unsigned)(unsigned long long)l;  // LDS byte offset
  asm volatile("global_load_async_to_lds_b128 %0, %1, off"
               :: "v"(loff), "v"(g)
               : "memory");
}

// ---------------- Kernel 0: fp32 -> f16 convert ----------------
__global__ void cvt_f32_f16(const float* __restrict__ s,
                            _Float16* __restrict__ d, int n) {
  int i = blockIdx.x * blockDim.x + threadIdx.x;
  const int stride = gridDim.x * blockDim.x;
  for (; i < n; i += stride) d[i] = (_Float16)s[i];
}

// ---------------- Kernel 1: fused QKV projection ----------------
// Y = xh @ W^T + b, scattered to per-head layouts:
//   Q,K -> [bh][s][hd]   (row-major, pitch 64)
//   V   -> [bh][hd][s]   (transposed, pitch 2048)
// Block = 256 threads = 8 waves as 4(M) x 2(N); block tile 128x128;
// wave tile 32x64 => 8 v_wmma per 32-deep K step, 128 per wave total.
// Direct-global operand loads: compiler double-buffers these well and the
// 0.5 MB weights live in the 192 MB L2.
__global__ __launch_bounds__(256) void qkv_kernel(
    const _Float16* __restrict__ xh, const _Float16* __restrict__ wh,
    const float* __restrict__ bq, const float* __restrict__ bk,
    const float* __restrict__ bv,
    _Float16* __restrict__ qh, _Float16* __restrict__ kh,
    _Float16* __restrict__ vt) {
  const int tid = threadIdx.x;
  const int lane = tid & 31;
  const int wid = tid >> 5;
  const int g = lane >> 4, cl = lane & 15;
  const int z = blockIdx.z;  // 0=Q, 1=K, 2=V
  const int m0 = blockIdx.x * 128 + (wid & 3) * 32;
  const int n0 = blockIdx.y * 128 + (wid >> 2) * 64;
  const _Float16* W = wh + (size_t)z * 512 * 512;  // row-major (n, k)
  const float* bias = (z == 0) ? bq : (z == 1) ? bk : bv;

  v8f acc[2][4] = {};
  for (int kk = 0; kk < 512; kk += 32) {
    v16h a0 = load_frag(xh, 512, m0, kk, lane);
    v16h a1 = load_frag(xh, 512, m0 + 16, kk, lane);
#pragma unroll
    for (int nt = 0; nt < 4; ++nt) {
      // B[k][n] = W[n][k]: B column n is W row n -> same packing as A.
      v16h bf = load_frag(W, 512, n0 + nt * 16, kk, lane);
      acc[0][nt] = wmma16(a0, bf, acc[0][nt]);
      acc[1][nt] = wmma16(a1, bf, acc[1][nt]);
    }
  }
#pragma unroll
  for (int nt = 0; nt < 4; ++nt) {
    const int n = n0 + nt * 16 + cl;
    const float bn = bias[n];
    const int h = n >> 6, hd = n & 63;
#pragma unroll
    for (int i = 0; i < 2; ++i) {
#pragma unroll
      for (int r = 0; r < 8; ++r) {
        const int m = m0 + i * 16 + r + 8 * g;  // D layout: row = r + 8g
        const int b = m >> 11, s = m & 2047;
        const _Float16 val = (_Float16)(acc[i][nt][r] + bn);
        if (z == 2) {
          vt[((size_t)(b * 8 + h) * 64 + hd) * 2048 + s] = val;
        } else {
          _Float16* dst = (z == 0) ? qh : kh;
          dst[((size_t)(b * 8 + h) * 2048 + s) * 64 + hd] = val;
        }
      }
    }
  }
}

// ---------------- Kernel 2: flash attention ----------------
// Block = 8 waves handling 8 consecutive 16-query tiles of ONE (b,h); all
// waves share the same K/V stream. Per 32-key iteration the block stages the
// K tile (32x64 f16) and V tile (64x32 f16) into LDS with async loads
// (double-buffered, ASYNCcnt), then each wave:
//   scores via 4 WMMAs, online softmax (shfl reductions + v_exp),
//   P relayout D->A through per-wave LDS, P@V via 4 WMMAs.
__global__ __launch_bounds__(256) void attn_kernel(
    const _Float16* __restrict__ qh, const _Float16* __restrict__ kh,
    const _Float16* __restrict__ vt, _Float16* __restrict__ ch) {
  // Padded pitches (72 / 40 halves) keep 16B ds reads spread across banks.
  __shared__ __align__(16) _Float16 sK[2][32 * 72];  // [key][hd]
  __shared__ __align__(16) _Float16 sV[2][64 * 40];  // [hd][key]
  __shared__ __align__(16) _Float16 sP[8][16 * 32];  // per-wave P relayout
  const int tid = threadIdx.x, lane = tid & 31, wid = tid >> 5;
  const int g = lane >> 4, cl = lane & 15;
  const int w = blockIdx.x * 8 + wid;    // 4096 wave tasks
  const int bh = w >> 7;                 // 32 (b,h) pairs; uniform per block
  const int q0 = (w & 127) * 16;         // 128 query tiles per (b,h)
  const _Float16* Q = qh + (size_t)bh * 2048 * 64;
  const _Float16* K = kh + (size_t)bh * 2048 * 64;
  const _Float16* V = vt + (size_t)bh * 64 * 2048;   // [hd][s]

  const v16h aq0 = load_frag(Q, 64, q0, 0, lane);
  const v16h aq1 = load_frag(Q, 64, q0, 32, lane);

  v8f acc[4] = {};
  float mrow[8], lrow[8];
#pragma unroll
  for (int r = 0; r < 8; ++r) { mrow[r] = -3.0e38f; lrow[r] = 0.0f; }

  // Cooperative stage of one 32-key K/V tile pair: 2 async b128 per thread.
  const int krow = tid >> 3, kcol = (tid & 7) * 8;   // K: 32 rows x 64 halves
  const int vrow = tid >> 2, vcol = (tid & 3) * 8;   // V: 64 rows x 32 halves
  auto stage = [&](int buf, int key0) {
    async_b128(K + (size_t)(key0 + krow) * 64 + kcol,
               &sK[buf][krow * 72 + kcol]);
    async_b128(V + (size_t)vrow * 2048 + key0 + vcol,
               &sV[buf][vrow * 40 + vcol]);
  };

  stage(0, 0);
  for (int kb = 0; kb < 64; ++kb) {
    const int cur = kb & 1;
    if (kb + 1 < 64) {
      stage(cur ^ 1, (kb + 1) * 32);
      asm volatile("s_wait_asynccnt 0x2" ::: "memory");  // cur tiles landed
    } else {
      asm volatile("s_wait_asynccnt 0x0" ::: "memory");
    }
    __syncthreads();  // all waves' copies visible

    const _Float16* tK = &sK[cur][0];
    const _Float16* tV = &sV[cur][0];
    v8f s0 = {}, s1 = {};
    {
      // K rows (local keys) are the B-matrix columns of Q.K^T.
      v16h b00 = load_frag(tK, 72, 0, 0, lane);
      v16h b01 = load_frag(tK, 72, 0, 32, lane);
      s0 = wmma16(aq0, b00, s0);
      s0 = wmma16(aq1, b01, s0);
      v16h b10 = load_frag(tK, 72, 16, 0, lane);
      v16h b11 = load_frag(tK, 72, 16, 32, lane);
      s1 = wmma16(aq0, b10, s1);
      s1 = wmma16(aq1, b11, s1);
    }
#pragma unroll
    for (int r = 0; r < 8; ++r) {
      const float v0 = s0[r] * 0.125f;   // 1/sqrt(64)
      const float v1 = s1[r] * 0.125f;
      const float mx = rmax16(fmaxf(v0, v1));
      const float mn = fmaxf(mrow[r], mx);
      const float corr = __expf(mrow[r] - mn);
      const float p0 = __expf(v0 - mn);
      const float p1 = __expf(v1 - mn);
      lrow[r] = lrow[r] * corr + rsum16(p0 + p1);
      mrow[r] = mn;
      acc[0][r] *= corr; acc[1][r] *= corr;
      acc[2][r] *= corr; acc[3][r] *= corr;
      const int row = r + 8 * g;         // D layout -> LDS row-major 16x32
      sP[wid][row * 32 + cl]      = (_Float16)p0;
      sP[wid][row * 32 + 16 + cl] = (_Float16)p1;
    }
    asm volatile("s_wait_dscnt 0x0" ::: "memory");  // P stores visible (same wave)
    const v16h pf = load_frag(&sP[wid][0], 32, 0, 0, lane);  // A fragment
#pragma unroll
    for (int nt = 0; nt < 4; ++nt) {
      // B[k][n] = V[key][hd] = sV[hd][key]: sV rows are B columns.
      v16h bvf = load_frag(tV, 40, nt * 16, 0, lane);
      acc[nt] = wmma16(pf, bvf, acc[nt]);
    }
    __syncthreads();  // reads of cur done before it is re-staged at kb+2
  }
  const int b = bh >> 3, h = bh & 7;
#pragma unroll
  for (int r = 0; r < 8; ++r) {
    const float inv = 1.0f / lrow[r];
    const size_t rowoff = ((size_t)b * 2048 + (q0 + r + 8 * g)) * 512 + h * 64;
#pragma unroll
    for (int nt = 0; nt < 4; ++nt)
      ch[rowoff + nt * 16 + cl] = (_Float16)(acc[nt][r] * inv);
  }
}

// ---------------- Kernel 3: output projection ----------------
__global__ __launch_bounds__(256) void oproj_kernel(
    const _Float16* __restrict__ chx, const _Float16* __restrict__ wo,
    const float* __restrict__ bo, float* __restrict__ out) {
  const int tid = threadIdx.x, lane = tid & 31, wid = tid >> 5;
  const int g = lane >> 4, cl = lane & 15;
  const int m0 = blockIdx.x * 128 + (wid & 3) * 32;
  const int n0 = blockIdx.y * 128 + (wid >> 2) * 64;
  v8f acc[2][4] = {};
  for (int kk = 0; kk < 512; kk += 32) {
    v16h a0 = load_frag(chx, 512, m0, kk, lane);
    v16h a1 = load_frag(chx, 512, m0 + 16, kk, lane);
#pragma unroll
    for (int nt = 0; nt < 4; ++nt) {
      v16h bf = load_frag(wo, 512, n0 + nt * 16, kk, lane);
      acc[0][nt] = wmma16(a0, bf, acc[0][nt]);
      acc[1][nt] = wmma16(a1, bf, acc[1][nt]);
    }
  }
#pragma unroll
  for (int nt = 0; nt < 4; ++nt) {
    const int n = n0 + nt * 16 + cl;
    const float bn = bo[n];
#pragma unroll
    for (int i = 0; i < 2; ++i) {
#pragma unroll
      for (int r = 0; r < 8; ++r) {
        out[(size_t)(m0 + i * 16 + r + 8 * g) * 512 + n] = acc[i][nt][r] + bn;
      }
    }
  }
}

// ---------------- launch ----------------
extern "C" void kernel_launch(void* const* d_in, const int* in_sizes, int n_in,
                              void* d_out, int out_size, void* d_ws,
                              size_t ws_size, hipStream_t stream) {
  (void)in_sizes; (void)n_in; (void)out_size; (void)ws_size;
  const float* x  = (const float*)d_in[0];
  const float* Wq = (const float*)d_in[1];
  const float* bq = (const float*)d_in[2];
  const float* Wk = (const float*)d_in[3];
  const float* bk = (const float*)d_in[4];
  const float* Wv = (const float*)d_in[5];
  const float* bv = (const float*)d_in[6];
  const float* Wo = (const float*)d_in[7];
  const float* bo = (const float*)d_in[8];

  char* ws = (char*)d_ws;
  _Float16* xh = (_Float16*)(ws + 0);                 //  8 MB: x in f16
  _Float16* wh = (_Float16*)(ws + (8u << 20));        //  2 MB: Wq,Wk,Wv,Wo f16
  _Float16* qh = (_Float16*)(ws + 10485760u);         //  8 MB: Q [bh][s][hd]
  _Float16* kh = (_Float16*)(ws + 18874368u);         //  8 MB: K [bh][s][hd]
  _Float16* vt = (_Float16*)(ws + 27262976u);         //  8 MB: V [bh][hd][s]
  _Float16* chx = (_Float16*)(ws + 35651584u);        //  8 MB: ctx [b][s][h*hd]

  const int NW = 512 * 512;
  cvt_f32_f16<<<4096, 256, 0, stream>>>(x, xh, 8192 * 512);
  cvt_f32_f16<<<256, 256, 0, stream>>>(Wq, wh + 0 * (size_t)NW, NW);
  cvt_f32_f16<<<256, 256, 0, stream>>>(Wk, wh + 1 * (size_t)NW, NW);
  cvt_f32_f16<<<256, 256, 0, stream>>>(Wv, wh + 2 * (size_t)NW, NW);
  cvt_f32_f16<<<256, 256, 0, stream>>>(Wo, wh + 3 * (size_t)NW, NW);

  qkv_kernel<<<dim3(64, 4, 3), 256, 0, stream>>>(xh, wh, bq, bk, bv,
                                                 qh, kh, vt);
  attn_kernel<<<512, 256, 0, stream>>>(qh, kh, vt, chx);
  oproj_kernel<<<dim3(64, 4), 256, 0, stream>>>(chx, wh + 3 * (size_t)NW,
                                                bo, (float*)d_out);
}
